// GCN_26225070309437
// MI455X (gfx1250) — matrix operand order repaired and checked by reading, not compiled, not run on hardware
//
#include <hip/hip_runtime.h>

typedef __attribute__((ext_vector_type(16))) __bf16 v16bf;
typedef __attribute__((ext_vector_type(8)))  float  v8f;

#define FEAT_K 128

// Pack two f32 into one dword of bf16 (lo = a, hi = b) using native converts.
__device__ __forceinline__ unsigned int pack2bf(float a, float b) {
  union { unsigned int u; __bf16 h[2]; } p;
  p.h[0] = (__bf16)a;
  p.h[1] = (__bf16)b;
  return p.u;
}

union Frag16 { v16bf v; unsigned int d[8]; };

// ---------------------------------------------------------------------------
// GEMM: H[nrows x nout] = X[nrows x 128] @ W[128 x nout], bf16 WMMA, f32 accum
// Workgroup = 256 threads = 8 waves; wave w owns rows [blk*128 + w*16, +16)
// and all NT column tiles. W pre-swizzled into LDS in B-fragment layout.
// ---------------------------------------------------------------------------
template<int NT>
__global__ __launch_bounds__(256) void gcn_gemm_wmma(
    const float* __restrict__ X, const float* __restrict__ W,
    float* __restrict__ H, int nrows, int nout) {
  __shared__ unsigned int ldsW[NT * 4 * 32 * 8];   // NT tiles * 4 ksteps * 32 lanes * 8 dwords
  const int tid = threadIdx.x;

  // Pack W (f32, row-major K x nout) into bf16 B-fragments:
  // frag(ks,tn), lane L, dword j holds W[ks*32+(L/16)*8+2j+(j/4)*8, tn*16+L%16] (lo)
  // and row+1 (hi).  Zero-pad columns >= nout.
  const int total = NT * 4 * 32 * 8;
  for (int idx = tid; idx < total; idx += 256) {
    int j  = idx & 7;
    int L  = (idx >> 3) & 31;
    int t  = idx >> 8;
    int tn = t % NT;
    int ks = t / NT;
    int k  = ks * 32 + (L >> 4) * 8 + 2 * j + (j >> 2) * 8;
    int n  = tn * 16 + (L & 15);
    float w0 = (n < nout) ? W[k * nout + n]       : 0.0f;
    float w1 = (n < nout) ? W[(k + 1) * nout + n] : 0.0f;
    ldsW[idx] = pack2bf(w0, w1);
  }
  __syncthreads();

  const int lane = tid & 31;
  const int wave = tid >> 5;
  const int g    = lane >> 4;
  const int row0 = blockIdx.x * 128 + wave * 16;
  int m  = row0 + (lane & 15);
  int mi = m < nrows ? m : nrows - 1;              // clamp loads; stores guarded
  const float* xrow = X + (size_t)mi * FEAT_K;

  v8f acc[NT] = {};

#pragma unroll
  for (int ks = 0; ks < 4; ++ks) {
    // A fragment 16x32 bf16: elems 0-7 = K(ks*32+g*8 ..), elems 8-15 = +16
    Frag16 A;
    const float* xp = xrow + ks * 32 + g * 8;
#pragma unroll
    for (int i = 0; i < 4; ++i) A.d[i]     = pack2bf(xp[2*i],      xp[2*i+1]);
#pragma unroll
    for (int i = 0; i < 4; ++i) A.d[4 + i] = pack2bf(xp[16 + 2*i], xp[16 + 2*i + 1]);

#pragma unroll
    for (int tn = 0; tn < NT; ++tn) {
      Frag16 B;
      const uint4* bp = (const uint4*)&ldsW[(((ks * NT + tn) * 32 + lane) << 3)];
      uint4 q0 = bp[0], q1 = bp[1];
      B.d[0] = q0.x; B.d[1] = q0.y; B.d[2] = q0.z; B.d[3] = q0.w;
      B.d[4] = q1.x; B.d[5] = q1.y; B.d[6] = q1.z; B.d[7] = q1.w;
      acc[tn] = __builtin_amdgcn_wmma_f32_16x16x32_bf16(
          false, A.v, false, B.v, (short)0, acc[tn], false, false);
    }
  }

  // C layout: VGPR r, lane L -> row (L/16)*8 + r, col L%16
#pragma unroll
  for (int tn = 0; tn < NT; ++tn) {
#pragma unroll
    for (int r = 0; r < 8; ++r) {
      int mr = row0 + g * 8 + r;
      int n  = tn * 16 + (lane & 15);
      if (mr < nrows && n < nout) H[(size_t)mr * nout + n] = acc[tn][r];
    }
  }
}

// ---------------------------------------------------------------------------
// Zero fill (float4-wide; n4 = count of float4 elements)
// ---------------------------------------------------------------------------
__global__ void gcn_fill_zero4(float4* __restrict__ p, size_t n4) {
  size_t i = (size_t)blockIdx.x * blockDim.x + threadIdx.x;
  if (i < n4) p[i] = make_float4(0.f, 0.f, 0.f, 0.f);
}

// ---------------------------------------------------------------------------
// Degree: deg[dst[e]] += 1  (native f32 atomic)
// ---------------------------------------------------------------------------
__global__ void gcn_degree(const int* __restrict__ dst, float* __restrict__ deg, int nE) {
  int e = blockIdx.x * blockDim.x + threadIdx.x;
  if (e < nE) unsafeAtomicAdd(&deg[dst[e]], 1.0f);
}

__global__ void gcn_dinv(float* __restrict__ deg, int n) {
  int i = blockIdx.x * blockDim.x + threadIdx.x;
  if (i < n) deg[i] = rsqrtf(1.0f + deg[i]);
}

// ---------------------------------------------------------------------------
// Edge scatter: AGG[dst,:] += H[src,:] * dinv[src]*dinv[dst]
// One wave per edge: coalesced 512B row gather + f32 atomic scatter (L2-resident)
// ---------------------------------------------------------------------------
__global__ __launch_bounds__(256) void gcn_edge_agg(
    const float* __restrict__ H, const int* __restrict__ src,
    const int* __restrict__ dst, const float* __restrict__ dinv,
    float* __restrict__ AGG, int nE, int nout) {
  int lane = threadIdx.x & 31;
  int e = blockIdx.x * 8 + (threadIdx.x >> 5);
  if (e >= nE) return;
  int s = src[e], d = dst[e];
  float norm = dinv[s] * dinv[d];
  const float4* hp = (const float4*)(H + (size_t)s * nout);
  float* ap = AGG + (size_t)d * nout;
  int nq = nout >> 2;
  for (int q = lane; q < nq; q += 32) {
    float4 v = hp[q];
    unsafeAtomicAdd(ap + 4 * q + 0, v.x * norm);
    unsafeAtomicAdd(ap + 4 * q + 1, v.y * norm);
    unsafeAtomicAdd(ap + 4 * q + 2, v.z * norm);
    unsafeAtomicAdd(ap + 4 * q + 3, v.w * norm);
  }
}

// ---------------------------------------------------------------------------
// Fused self-loop + bias + optional ReLU, in place on AGG.
// One block (128 threads) per node row: no div/mod, fully coalesced.
// ---------------------------------------------------------------------------
__global__ __launch_bounds__(128) void gcn_selfbias(
    float* __restrict__ AGG, const float* __restrict__ H,
    const float* __restrict__ dinv, const float* __restrict__ bias,
    int nout, int relu) {
  int row = blockIdx.x;
  int col = threadIdx.x;
  if (col >= nout) return;
  float di = dinv[row];
  size_t i = (size_t)row * nout + col;
  float v = AGG[i] + H[i] * di * di + bias[col];
  if (relu) v = fmaxf(v, 0.0f);
  AGG[i] = v;
}

// ---------------------------------------------------------------------------
// log_softmax over nout (<=64) columns; one wave per node
// ---------------------------------------------------------------------------
__global__ __launch_bounds__(256) void gcn_logsoftmax(
    const float* __restrict__ X, float* __restrict__ OUT, int nrows, int nout) {
  int lane = threadIdx.x & 31;
  int n = blockIdx.x * 8 + (threadIdx.x >> 5);
  if (n >= nrows) return;
  const float* xp = X + (size_t)n * nout;
  float v0 = (lane < nout)      ? xp[lane]      : -__builtin_inff();
  float v1 = (lane + 32 < nout) ? xp[lane + 32] : -__builtin_inff();
  float m = fmaxf(v0, v1);
#pragma unroll
  for (int off = 16; off; off >>= 1) m = fmaxf(m, __shfl_xor(m, off, 32));
  float s = ((lane < nout) ? expf(v0 - m) : 0.0f) +
            ((lane + 32 < nout) ? expf(v1 - m) : 0.0f);
#pragma unroll
  for (int off = 16; off; off >>= 1) s += __shfl_xor(s, off, 32);
  float lse = m + logf(s);
  if (lane < nout)      OUT[(size_t)n * nout + lane]      = v0 - lse;
  if (lane + 32 < nout) OUT[(size_t)n * nout + lane + 32] = v1 - lse;
}

// ---------------------------------------------------------------------------
extern "C" void kernel_launch(void* const* d_in, const int* in_sizes, int n_in,
                              void* d_out, int out_size, void* d_ws, size_t ws_size,
                              hipStream_t stream) {
  const float* x  = (const float*)d_in[0];
  const int*   ei = (const int*)  d_in[1];
  const float* W1 = (const float*)d_in[2];
  const float* b1 = (const float*)d_in[3];
  const float* W2 = (const float*)d_in[4];
  const float* b2 = (const float*)d_in[5];
  const float* W3 = (const float*)d_in[6];
  const float* b3 = (const float*)d_in[7];
  const int N = in_sizes[0] / 128;
  const int E = in_sizes[1] / 2;
  const int* src = ei;
  const int* dst = ei + E;

  // workspace: dinv[N] | bufA[N*128] | bufB[N*128]   (~103 MB)
  size_t off = ((size_t)N * sizeof(float) + 255) & ~(size_t)255;
  float* dinv = (float*)d_ws;
  float* bufA = (float*)((char*)d_ws + off);
  float* bufB = bufA + (size_t)N * 128;

  dim3 blk(256);
  auto nblk = [](size_t work, int per) { return dim3((unsigned)((work + per - 1) / per)); };
  const dim3 gemmGrid((N + 127) / 128);
  const size_t feat4  = (size_t)N * 128 / 4;     // float4 count, N*128 divisible by 4
  const size_t out4   = (size_t)N * 40 / 4;
  const size_t deg4   = ((size_t)N + 3) / 4;     // dinv region padded; safe (off >= N*4 rounded up)

  // degree + normalization
  gcn_fill_zero4<<<nblk(deg4, 256), blk, 0, stream>>>((float4*)dinv, deg4);
  gcn_degree    <<<nblk((size_t)E, 256), blk, 0, stream>>>(dst, dinv, E);
  gcn_dinv      <<<nblk((size_t)N, 256), blk, 0, stream>>>(dinv, N);

  // ---- layer 1 (128 -> 128, relu) ----
  gcn_gemm_wmma<8><<<gemmGrid, blk, 0, stream>>>(x, W1, bufA, N, 128);
  gcn_fill_zero4<<<nblk(feat4, 256), blk, 0, stream>>>((float4*)bufB, feat4);
  gcn_edge_agg  <<<nblk((size_t)E, 8), blk, 0, stream>>>(bufA, src, dst, dinv, bufB, E, 128);
  gcn_selfbias  <<<dim3(N), dim3(128), 0, stream>>>(bufB, bufA, dinv, b1, 128, 1);

  // ---- layer 2 (128 -> 128, relu) ----
  gcn_gemm_wmma<8><<<gemmGrid, blk, 0, stream>>>(bufB, W2, bufA, N, 128);
  gcn_fill_zero4<<<nblk(feat4, 256), blk, 0, stream>>>((float4*)bufB, feat4);
  gcn_edge_agg  <<<nblk((size_t)E, 8), blk, 0, stream>>>(bufA, src, dst, dinv, bufB, E, 128);
  gcn_selfbias  <<<dim3(N), dim3(128), 0, stream>>>(bufB, bufA, dinv, b2, 128, 1);

  // ---- layer 3 (128 -> 40) + log_softmax ----
  gcn_gemm_wmma<3><<<gemmGrid, blk, 0, stream>>>(bufB, W3, bufA, N, 40);
  gcn_fill_zero4<<<nblk(out4, 256), blk, 0, stream>>>((float4*)bufB, out4);
  gcn_edge_agg  <<<nblk((size_t)E, 8), blk, 0, stream>>>(bufA, src, dst, dinv, bufB, E, 40);
  gcn_selfbias  <<<dim3(N), dim3(128), 0, stream>>>(bufB, bufA, dinv, b3, 40, 0);
  gcn_logsoftmax<<<nblk((size_t)N, 8), blk, 0, stream>>>(bufB, (float*)d_out, N, 40);
}
